// AssemblyClassifier_69080253989006
// MI455X (gfx1250) — matrix operand order repaired and checked by reading, not compiled, not run on hardware
//
#include <hip/hip_runtime.h>
#include <hip/hip_bf16.h>

typedef __attribute__((ext_vector_type(16))) _Float16 v16h;
typedef __attribute__((ext_vector_type(8)))  float    v8f;
typedef __attribute__((ext_vector_type(4)))  float    v4f;   // native vec for NT loads

#define Bdim 8
#define Edim 28
#define Tdim 16384
#define Fdim 8
#define Adim 1024
#define Cdim 256
#define KPAD 64   // 2*Edim padded to 64 for two K=32 f16 WMMAs

// ---------------------------------------------------------------------------
// Kernel 1: fold (edge_present, eq_classes, scale, alpha) into transposed
// f16 weights Wt[c][k], k<28: -scale*G_edge[e=k][c], 28<=k<56: alpha*G_no,
// 56..63: zero pad.  grid = Edim blocks (one edge per block), 256 threads (c).
// ---------------------------------------------------------------------------
__global__ void __launch_bounds__(256) ac_prep_weights(
    const float* __restrict__ eq,          // (A,C)
    const unsigned char* __restrict__ ep,  // (A,E) bool bytes
    const float* __restrict__ scale_p,
    const float* __restrict__ alpha_p,
    _Float16* __restrict__ wt)             // (C, 64) f16
{
  __shared__ unsigned char efs[Adim];
  const int c = threadIdx.x;
  const int e = blockIdx.x;
  for (int a = c; a < Adim; a += 256) efs[a] = ep[a * Edim + e];
  __syncthreads();

  float accE = 0.f, csum = 0.f;
  for (int a = 0; a < Adim; ++a) {
    float v = eq[a * Cdim + c];       // coalesced row read across threads
    csum += v;
    if (efs[a]) accE += v;
  }
  const float scale = scale_p[0];
  const float alpha = alpha_p[0];
  wt[c * KPAD + e]        = (_Float16)(-scale * accE);
  wt[c * KPAD + Edim + e] = (_Float16)(alpha * (csum - accE));
  if (e == 0) {
#pragma unroll
    for (int k = 2 * Edim; k < KPAD; ++k) wt[c * KPAD + k] = (_Float16)0.f;
  }
}

// ---------------------------------------------------------------------------
// Kernel 2: streaming WMMA GEMM.  grid = B * (T/128) = 1024 blocks,
// 256 threads = 8 waves, each wave owns 16 rows (t) x all 256 cols (c).
// ---------------------------------------------------------------------------
__global__ void __launch_bounds__(256, 1) ac_main(
    const float* __restrict__ x,     // (B,E,T,F)
    const _Float16* __restrict__ wt, // (C, 64)
    float* __restrict__ out)         // (B,T,C)
{
  // 32KB weights + 8 waves * 2KB A-tiles = 48KB LDS
  __shared__ __align__(16) _Float16 lds[Cdim * KPAD + 8 * 16 * KPAD];

  const int tid  = threadIdx.x;
  const int lane = tid & 31;
  const int wv   = tid >> 5;
  const int b     = blockIdx.x >> 7;           // 128 blocks per batch
  const int t_blk = (blockIdx.x & 127) * 128;  // 128 rows per block

  // ---- stage weights into LDS (2048 uint4 total, 8 per thread) ----
  {
    const uint4* src = (const uint4*)wt;
    uint4*       dst = (uint4*)lds;
#pragma unroll
    for (int i = 0; i < 8; ++i) dst[tid + i * 256] = src[tid + i * 256];
  }
  __syncthreads();

  const int t0 = t_blk + wv * 16;
  _Float16* aT = &lds[Cdim * KPAD + wv * 16 * KPAD];  // 16 x 64 f16 tile

  // ---- stream x, reduce features, NaN mask -> A tile in LDS ----
  // 16 rows * 28 edges = 448 (t,e) pairs; 14 per lane.
#pragma unroll
  for (int i = 0; i < 14; ++i) {
    const int p = lane + i * 32;
    const int t = p / Edim;
    const int e = p - t * Edim;
    const v4f* xp =
        (const v4f*)x + (((size_t)b * Edim + e) * Tdim + (size_t)(t0 + t)) * 2;
    v4f lo = __builtin_nontemporal_load(xp);
    v4f hi = __builtin_nontemporal_load(xp + 1);
    float s = lo.x + lo.y + lo.z + lo.w + hi.x + hi.y + hi.z + hi.w;
    const bool obs = !__builtin_isnan(s);
    aT[t * KPAD + e]        = obs ? (_Float16)s : (_Float16)0.f;
    aT[t * KPAD + Edim + e] = obs ? (_Float16)1.f : (_Float16)0.f;
  }
  if (lane < 16) {  // zero K = 56..63 pad (16B store)
    uint4 z = {0u, 0u, 0u, 0u};
    *(uint4*)&aT[lane * KPAD + 2 * Edim] = z;
  }
  __syncthreads();

  const int half = lane >> 4;   // K-half selector
  const int l15  = lane & 15;

  // ---- gather A fragments per documented 16-bit A 16x32 layout ----
  // lane<16: K {0..7,16..23}; lane>=16: K {8..15,24..31}; row M = lane%16
  union { v16h v; uint4 q[2]; } a0, a1;
  {
    const int ka = half ? 8 : 0;
    const int kb = half ? 24 : 16;
    const _Float16* ar = &aT[l15 * KPAD];
    a0.q[0] = *(const uint4*)&ar[ka];
    a0.q[1] = *(const uint4*)&ar[kb];
    a1.q[0] = *(const uint4*)&ar[32 + ka];
    a1.q[1] = *(const uint4*)&ar[32 + kb];
  }

  float* obase = out + ((size_t)b * Tdim + t0) * Cdim + l15;

#pragma unroll 4
  for (int n = 0; n < 16; ++n) {
    // B 32x16 f16 layout: lane<16 holds K {0..7,16..23} of column N=lane,
    // lane>=16 holds K {8..15,24..31} of column N=lane-16.  Wt is (c,k) so rows
    // are contiguous -> two ds_load_b128 per fragment.
    union { v16h v; uint4 q[2]; } b0, b1;
    const _Float16* wrow = &lds[(n * 16 + l15) * KPAD + half * 8];
    b0.q[0] = *(const uint4*)&wrow[0];
    b0.q[1] = *(const uint4*)&wrow[16];
    b1.q[0] = *(const uint4*)&wrow[32];
    b1.q[1] = *(const uint4*)&wrow[48];

    v8f acc = {};
    acc = __builtin_amdgcn_wmma_f32_16x16x32_f16(false, a0.v, false, b0.v,
                                                 (short)0, acc, false, false);
    acc = __builtin_amdgcn_wmma_f32_16x16x32_f16(false, a1.v, false, b1.v,
                                                 (short)0, acc, false, false);

    // D layout: VGPR r -> M = r + 8*half, N = lane%16
    float* op = obase + n * 16 + (size_t)(half * 8) * Cdim;
#pragma unroll
    for (int r = 0; r < 8; ++r)
      __builtin_nontemporal_store(acc[r], op + (size_t)r * Cdim);
  }
}

extern "C" void kernel_launch(void* const* d_in, const int* in_sizes, int n_in,
                              void* d_out, int out_size, void* d_ws, size_t ws_size,
                              hipStream_t stream) {
  const float*         x     = (const float*)d_in[0];          // (B,E,T,F)
  const float*         eq    = (const float*)d_in[1];          // (A,C)
  const float*         scale = (const float*)d_in[2];          // scalar
  const float*         alpha = (const float*)d_in[3];          // scalar
  const unsigned char* ep    = (const unsigned char*)d_in[4];  // (A,E) bool
  _Float16* wt  = (_Float16*)d_ws;                             // 32KB scratch
  float*    out = (float*)d_out;                               // (B,T,C)

  ac_prep_weights<<<Edim, 256, 0, stream>>>(eq, ep, scale, alpha, wt);
  ac_main<<<Bdim * (Tdim / 128), 256, 0, stream>>>(x, wt, out);
}
